// GroupedQueryAttention_17291538333752
// MI455X (gfx1250) — matrix operand order repaired and checked by reading, not compiled, not run on hardware
//
#include <hip/hip_runtime.h>
#include <hip/hip_bf16.h>
#include <cstdint>

// ---------------- types ----------------
typedef __bf16 bf16;
typedef bf16  v16bf __attribute__((ext_vector_type(16)));
typedef bf16  v8bf  __attribute__((ext_vector_type(8)));
typedef float v8f   __attribute__((ext_vector_type(8)));

// ---------------- problem constants ----------------
#define BATCH 2
#define SEQ   2048
#define DIM   2048
#define NHEAD 16
#define NGRP  4
#define HDIM  128
#define KVD   (NGRP * HDIM)   // 512
#define MROWS (BATCH * SEQ)   // 4096
#define SCALE 0.08838834764831845f  // 1/sqrt(128)
#define NEG_INF (-3.0e38f)

// ---------------- helpers ----------------
__device__ __forceinline__ unsigned short f2bf(float f) {
    union { float f; unsigned int u; } x; x.f = f;
    unsigned int u = x.u;
    unsigned int r = (u + 0x7fffu + ((u >> 16) & 1u)) >> 16;
    return (unsigned short)r;
}

__device__ __forceinline__ v16bf ldfrag(const unsigned short* p0, const unsigned short* p1) {
    v16bf r;
    ((v8bf*)&r)[0] = *(const v8bf*)p0;
    ((v8bf*)&r)[1] = *(const v8bf*)p1;
    return r;
}

__device__ __forceinline__ v8f wmma_bf16(v16bf a, v16bf b, v8f c) {
    return __builtin_amdgcn_wmma_f32_16x16x32_bf16(false, a, false, b, (short)0, c, false, false);
}

// async global->LDS 16B copy, saddr form: per-lane LDS byte addr + per-lane
// 32-bit byte offset + uniform 64-bit base in SGPRs. Tracked by ASYNCcnt.
__device__ __forceinline__ void async_ld_b128(unsigned lds_addr, unsigned goff,
                                              const unsigned short* base) {
    asm volatile("global_load_async_to_lds_b128 %0, %1, %2"
                 :: "v"(lds_addr), "v"(goff), "s"(base) : "memory");
}

// ---------------- conversion kernels ----------------
__global__ void cvt_f32_to_bf16(const float* __restrict__ in, unsigned short* __restrict__ out, size_t n) {
    size_t i = (size_t)blockIdx.x * blockDim.x + threadIdx.x;
    size_t stride = (size_t)gridDim.x * blockDim.x;
    for (; i < n; i += stride) out[i] = f2bf(in[i]);
}

// in: (K, N) f32 row-major  ->  out: (N, K) bf16 row-major
__global__ void cvt_transpose_bf16(const float* __restrict__ in, unsigned short* __restrict__ out,
                                   int K, int N) {
    size_t n_total = (size_t)K * N;
    size_t i = (size_t)blockIdx.x * blockDim.x + threadIdx.x;
    size_t stride = (size_t)gridDim.x * blockDim.x;
    for (; i < n_total; i += stride) {
        int k = (int)(i / N);
        int n = (int)(i % N);
        out[(size_t)n * K + k] = f2bf(in[i]);
    }
}

// ---------------- bf16 WMMA GEMM: C(f32)[M][N] = A[M][K] * Bt[N][K]^T ----------------
// BM=BN=128, BK=64, 256 threads (8 waves), each wave computes 32x64.
// Double-buffered LDS, tiles staged with GLOBAL_LOAD_ASYNC_TO_LDS_B128.
#define LDT 72                       // padded LDS stride in halves (144B, 16B aligned)
#define TILE_HALVES (128 * LDT)      // one buffer, in halves
#define TILE_BYTES  (TILE_HALVES * 2)

__global__ __launch_bounds__(256) void gemm_bf16_f32(const unsigned short* __restrict__ A,
                                                     const unsigned short* __restrict__ Bt,
                                                     float* __restrict__ C,
                                                     int M, int N, int K) {
    __shared__ alignas(16) unsigned short sA[2][TILE_HALVES];
    __shared__ alignas(16) unsigned short sB[2][TILE_HALVES];
    const int tid  = threadIdx.x;
    const int wave = tid >> 5, lane = tid & 31;
    const int l15  = lane & 15, hi = lane >> 4;
    const int bm = blockIdx.y * 128, bn = blockIdx.x * 128;
    const int wm = (wave >> 1) * 32, wn = (wave & 1) * 64;

    // this thread's 4 staging chunks (8 halves = 16B each) per operand
    unsigned gA[4], gB[4], lA[4], lB[4];
#pragma unroll
    for (int c = 0; c < 4; ++c) {
        int idx = tid + c * 256;              // 1024 chunks per tile
        int row = idx >> 3;
        int kc  = (idx & 7) * 8;
        gA[c] = (unsigned)(((size_t)(bm + row) * K + kc) * 2);   // byte offset, k-tile 0
        gB[c] = (unsigned)(((size_t)(bn + row) * K + kc) * 2);
        lA[c] = (unsigned)(size_t)&sA[0][row * LDT + kc];        // LDS byte addr, buffer 0
        lB[c] = (unsigned)(size_t)&sB[0][row * LDT + kc];
    }

    v8f acc[2][4] = {};
    const int nt = K >> 6;     // number of 64-wide k tiles

    // prologue: stage tile 0 into buffer 0
#pragma unroll
    for (int c = 0; c < 4; ++c) {
        async_ld_b128(lA[c], gA[c], A);
        async_ld_b128(lB[c], gB[c], Bt);
    }

    for (int kt = 0; kt < nt; ++kt) {
        const int cur = kt & 1, nxt = cur ^ 1;
        __syncthreads();                       // everyone done computing from `nxt`
        if (kt + 1 < nt) {
            unsigned kb = (unsigned)((kt + 1) * 128);   // 64 halves = 128 bytes
#pragma unroll
            for (int c = 0; c < 4; ++c) {
                async_ld_b128(lA[c] + nxt * TILE_BYTES, gA[c] + kb, A);
                async_ld_b128(lB[c] + nxt * TILE_BYTES, gB[c] + kb, Bt);
            }
            asm volatile("s_wait_asynccnt 0x8" ::: "memory");  // tile kt landed
        } else {
            asm volatile("s_wait_asynccnt 0x0" ::: "memory");
        }
        __syncthreads();                       // all waves' slices of tile kt visible

        const unsigned short* sAc = sA[cur];
        const unsigned short* sBc = sB[cur];
#pragma unroll
        for (int ks = 0; ks < 2; ++ks) {
            v16bf af[2], bfr[4];
#pragma unroll
            for (int s = 0; s < 2; ++s) {
                const unsigned short* p = sAc + (wm + s * 16 + l15) * LDT + ks * 32;
                af[s] = ldfrag(p + hi * 8, p + 16 + hi * 8);
            }
#pragma unroll
            for (int j = 0; j < 4; ++j) {
                const unsigned short* p = sBc + (wn + j * 16 + l15) * LDT + ks * 32;
                bfr[j] = ldfrag(p + hi * 8, p + 16 + hi * 8);
            }
#pragma unroll
            for (int s = 0; s < 2; ++s)
#pragma unroll
                for (int j = 0; j < 4; ++j)
                    acc[s][j] = wmma_bf16(af[s], bfr[j], acc[s][j]);
        }
    }
#pragma unroll
    for (int s = 0; s < 2; ++s)
#pragma unroll
        for (int j = 0; j < 4; ++j)
#pragma unroll
            for (int vg = 0; vg < 8; ++vg) {
                int row = bm + wm + s * 16 + vg + hi * 8;
                int col = bn + wn + j * 16 + l15;
                C[(size_t)row * N + col] = acc[s][j][vg];
            }
}

// ---------------- RoPE + scatter ----------------
// qf: (B*S, D) f32 ; kf,vf: (B*S, 512) f32 ; cos/sin: (S,64)
// qb16: (B,H,S,HD) bf16 ; kb16/vb16: (B,G,S,HD) bf16 ; kcache/vcache: (B,G,S,HD) f32
__global__ __launch_bounds__(256) void rope_scatter(const float* __restrict__ qf,
                                                    const float* __restrict__ kf,
                                                    const float* __restrict__ vf,
                                                    const float* __restrict__ cosp,
                                                    const float* __restrict__ sinp,
                                                    unsigned short* __restrict__ qb16,
                                                    unsigned short* __restrict__ kb16,
                                                    unsigned short* __restrict__ vb16,
                                                    float* __restrict__ kcache,
                                                    float* __restrict__ vcache) {
    const int bs = blockIdx.x;             // 0..B*S-1
    const int b  = bs / SEQ, s = bs % SEQ;
    const int tid = threadIdx.x;
    const float* cr = cosp + (size_t)s * 64;
    const float* sr = sinp + (size_t)s * 64;

    // Q: 1024 rotary pairs
    for (int p = tid; p < NHEAD * 64; p += 256) {
        int h = p >> 6, j = p & 63;
        size_t in = (size_t)bs * DIM + h * HDIM + 2 * j;
        float x1 = qf[in], x2 = qf[in + 1];
        float c = cr[j], sn = sr[j];
        float r1 = x1 * c - x2 * sn;
        float r2 = x1 * sn + x2 * c;
        size_t o = ((size_t)(b * NHEAD + h) * SEQ + s) * HDIM + 2 * j;
        qb16[o] = f2bf(r1); qb16[o + 1] = f2bf(r2);
    }
    // K: 256 rotary pairs (+ f32 cache)
    for (int p = tid; p < NGRP * 64; p += 256) {
        int g = p >> 6, j = p & 63;
        size_t in = (size_t)bs * KVD + g * HDIM + 2 * j;
        float x1 = kf[in], x2 = kf[in + 1];
        float c = cr[j], sn = sr[j];
        float r1 = x1 * c - x2 * sn;
        float r2 = x1 * sn + x2 * c;
        size_t o = ((size_t)(b * NGRP + g) * SEQ + s) * HDIM + 2 * j;
        kb16[o] = f2bf(r1); kb16[o + 1] = f2bf(r2);
        kcache[o] = r1; kcache[o + 1] = r2;
    }
    // V: 512 passthrough elements (+ f32 cache)
    for (int e = tid; e < KVD; e += 256) {
        int g = e >> 7, d = e & 127;
        float v = vf[(size_t)bs * KVD + e];
        size_t o = ((size_t)(b * NGRP + g) * SEQ + s) * HDIM + d;
        vb16[o] = f2bf(v); vcache[o] = v;
    }
}

// ---------------- Flash attention (causal, GQA) ----------------
// grid: (S/128, B*H), 256 threads. Q tile 128 rows; each wave owns 16 rows.
#define LDK 136  // 128 + 8 pad halves (272B, 16B aligned)

__global__ __launch_bounds__(256) void flash_attn(const unsigned short* __restrict__ qb16,
                                                  const unsigned short* __restrict__ kb16,
                                                  const unsigned short* __restrict__ vb16,
                                                  unsigned short* __restrict__ ctxb) {
    __shared__ alignas(16) unsigned short sK [128 * LDK];
    __shared__ alignas(16) unsigned short sVt[128 * LDK];
    __shared__ alignas(16) unsigned short sP [128 * LDK];   // 8 waves x 16 rows

    const int tid  = threadIdx.x;
    const int wave = tid >> 5, lane = tid & 31;
    const int l15  = lane & 15, hi = lane >> 4;
    const int qtile = blockIdx.x;
    const int q0    = qtile * 128;
    const int bh = blockIdx.y;
    const int b = bh / NHEAD, h = bh % NHEAD;
    const int g = h / (NHEAD / NGRP);

    const unsigned short* Q  = qb16 + ((size_t)(b * NHEAD + h) * SEQ + q0) * HDIM;
    const unsigned short* Kp = kb16 + (size_t)(b * NGRP + g) * SEQ * HDIM;
    const unsigned short* Vp = vb16 + (size_t)(b * NGRP + g) * SEQ * HDIM;

    // Q fragments for this wave's 16 rows, kept in registers (4 k-steps of 32)
    v16bf qfrag[4];
    {
        const unsigned short* qrow = Q + (size_t)(wave * 16 + l15) * HDIM;
#pragma unroll
        for (int kk = 0; kk < 4; ++kk)
            qfrag[kk] = ldfrag(qrow + kk * 32 + hi * 8, qrow + kk * 32 + 16 + hi * 8);
    }

    v8f o[8] = {};
    float mrow[8], lrow[8];
#pragma unroll
    for (int vg = 0; vg < 8; ++vg) { mrow[vg] = NEG_INF; lrow[vg] = 0.0f; }

    for (int kbk = 0; kbk <= qtile; ++kbk) {
        __syncthreads();
        // K tile: async DMA copy [key][hd] (8 x b128 per thread)
        for (int c = tid; c < 2048; c += 256) {
            int row = c >> 4, kc = (c & 15) * 8;
            unsigned lds = (unsigned)(size_t)&sK[row * LDK + kc];
            unsigned go  = (unsigned)(((size_t)(kbk * 128 + row) * HDIM + kc) * 2);
            async_ld_b128(lds, go, Kp);
        }
        // V tile: transposed copy -> [hd][key] (must pass through VGPRs)
        for (int c = tid; c < 2048; c += 256) {
            int row = c >> 4, kc = (c & 15) * 8;
            const unsigned short* vsrc = Vp + (size_t)(kbk * 128 + row) * HDIM + kc;
#pragma unroll
            for (int e = 0; e < 8; ++e) sVt[(kc + e) * LDK + row] = vsrc[e];
        }
        asm volatile("s_wait_asynccnt 0x0" ::: "memory");
        __syncthreads();

        // ---- S = Q K^T ----
        v8f sc[8];
        float mnew[8];
#pragma unroll
        for (int vg = 0; vg < 8; ++vg) mnew[vg] = mrow[vg];

        const int rowbase = q0 + wave * 16 + hi * 8;
#pragma unroll
        for (int j = 0; j < 8; ++j) {
            v8f sacc = {};
#pragma unroll
            for (int kk = 0; kk < 4; ++kk) {
                const unsigned short* p = sK + (j * 16 + l15) * LDK + kk * 32;
                v16bf kfrag = ldfrag(p + hi * 8, p + 16 + hi * 8);
                sacc = wmma_bf16(qfrag[kk], kfrag, sacc);
            }
            const int key = kbk * 128 + j * 16 + l15;
#pragma unroll
            for (int vg = 0; vg < 8; ++vg) {
                float val = sacc[vg] * SCALE;
                if (key > rowbase + vg) val = NEG_INF;   // causal mask
                sacc[vg] = val;
                mnew[vg] = fmaxf(mnew[vg], val);
            }
            sc[j] = sacc;
        }
        // row-max across the 16 lanes of each half-wave
#pragma unroll
        for (int vg = 0; vg < 8; ++vg) {
#pragma unroll
            for (int off = 1; off < 16; off <<= 1)
                mnew[vg] = fmaxf(mnew[vg], __shfl_xor(mnew[vg], off, 32));
        }
        float alpha[8], rsum[8];
#pragma unroll
        for (int vg = 0; vg < 8; ++vg) {
            alpha[vg] = __expf(mrow[vg] - mnew[vg]);
            mrow[vg]  = mnew[vg];
            rsum[vg]  = 0.0f;
        }
        // P = exp(S - m), write bf16 to per-wave LDS region
#pragma unroll
        for (int j = 0; j < 8; ++j) {
#pragma unroll
            for (int vg = 0; vg < 8; ++vg) {
                float pv = __expf(sc[j][vg] - mnew[vg]);
                rsum[vg] += pv;
                sP[(wave * 16 + vg + hi * 8) * LDK + j * 16 + l15] = f2bf(pv);
            }
        }
#pragma unroll
        for (int vg = 0; vg < 8; ++vg) {
#pragma unroll
            for (int off = 1; off < 16; off <<= 1)
                rsum[vg] += __shfl_xor(rsum[vg], off, 32);
            lrow[vg] = lrow[vg] * alpha[vg] + rsum[vg];
        }
        // rescale running O
#pragma unroll
        for (int t = 0; t < 8; ++t)
#pragma unroll
            for (int vg = 0; vg < 8; ++vg)
                o[t][vg] *= alpha[vg];

        // P was written cross-lane within this wave; wait for DS completion
        asm volatile("s_wait_dscnt 0" ::: "memory");

        // ---- O += P V ----
#pragma unroll
        for (int t = 0; t < 8; ++t) {
#pragma unroll
            for (int kk = 0; kk < 4; ++kk) {
                const unsigned short* pp = sP + (wave * 16 + l15) * LDK + kk * 32;
                v16bf pfrag = ldfrag(pp + hi * 8, pp + 16 + hi * 8);
                const unsigned short* vp = sVt + (t * 16 + l15) * LDK + kk * 32;
                v16bf vfrag = ldfrag(vp + hi * 8, vp + 16 + hi * 8);
                o[t] = wmma_bf16(pfrag, vfrag, o[t]);
            }
        }
    }

    // epilogue: O / l -> ctx bf16, layout (B, S, H*HD)
    float inv[8];
#pragma unroll
    for (int vg = 0; vg < 8; ++vg) inv[vg] = 1.0f / lrow[vg];
#pragma unroll
    for (int t = 0; t < 8; ++t)
#pragma unroll
        for (int vg = 0; vg < 8; ++vg) {
            int row = q0 + wave * 16 + vg + hi * 8;
            int col = h * HDIM + t * 16 + l15;
            ctxb[((size_t)b * SEQ + row) * DIM + col] = f2bf(o[t][vg] * inv[vg]);
        }
}

// ---------------- launch ----------------
extern "C" void kernel_launch(void* const* d_in, const int* in_sizes, int n_in,
                              void* d_out, int out_size, void* d_ws, size_t ws_size,
                              hipStream_t stream) {
    (void)in_sizes; (void)n_in; (void)out_size; (void)ws_size;
    const float* x    = (const float*)d_in[0];
    // d_in[1] = mask (bool) — causal triu(k=1), applied analytically
    const float* cosp = (const float*)d_in[2];
    const float* sinp = (const float*)d_in[3];
    const float* Wq   = (const float*)d_in[4];
    const float* Wk   = (const float*)d_in[5];
    const float* Wv   = (const float*)d_in[6];
    const float* Wo   = (const float*)d_in[7];

    float* out    = (float*)d_out;                       // (B,S,D)
    float* kcache = out + (size_t)MROWS * DIM;           // (B,G,S,HD)
    float* vcache = kcache + (size_t)BATCH * NGRP * SEQ * HDIM;

    // workspace partition (all offsets 256B aligned by construction)
    char* w = (char*)d_ws;
    size_t off = 0;
    unsigned short* xb   = (unsigned short*)(w + off); off += (size_t)MROWS * DIM * 2;        // 16M
    unsigned short* wqb  = (unsigned short*)(w + off); off += (size_t)DIM * DIM * 2;          // 8M
    unsigned short* wkb  = (unsigned short*)(w + off); off += (size_t)KVD * DIM * 2;          // 2M
    unsigned short* wvb  = (unsigned short*)(w + off); off += (size_t)KVD * DIM * 2;          // 2M
    unsigned short* wob  = (unsigned short*)(w + off); off += (size_t)DIM * DIM * 2;          // 8M
    float*          qf   = (float*)(w + off);          off += (size_t)MROWS * DIM * 4;        // 32M
    float*          kf   = (float*)(w + off);          off += (size_t)MROWS * KVD * 4;        // 8M
    float*          vf   = (float*)(w + off);          off += (size_t)MROWS * KVD * 4;        // 8M
    unsigned short* qb16 = (unsigned short*)(w + off); off += (size_t)MROWS * DIM * 2;        // 16M
    unsigned short* kb16 = (unsigned short*)(w + off); off += (size_t)MROWS * KVD * 2;        // 4M
    unsigned short* vb16 = (unsigned short*)(w + off); off += (size_t)MROWS * KVD * 2;        // 4M
    unsigned short* ctxb = (unsigned short*)(w + off); off += (size_t)MROWS * DIM * 2;        // 16M

    // 1) conversions
    cvt_f32_to_bf16<<<2048, 256, 0, stream>>>(x, xb, (size_t)MROWS * DIM);
    cvt_transpose_bf16<<<2048, 256, 0, stream>>>(Wq, wqb, DIM, DIM);
    cvt_transpose_bf16<<<1024, 256, 0, stream>>>(Wk, wkb, DIM, KVD);
    cvt_transpose_bf16<<<1024, 256, 0, stream>>>(Wv, wvb, DIM, KVD);
    cvt_transpose_bf16<<<2048, 256, 0, stream>>>(Wo, wob, DIM, DIM);

    // 2) Q/K/V projections
    gemm_bf16_f32<<<dim3(DIM / 128, MROWS / 128), 256, 0, stream>>>(xb, wqb, qf, MROWS, DIM, DIM);
    gemm_bf16_f32<<<dim3(KVD / 128, MROWS / 128), 256, 0, stream>>>(xb, wkb, kf, MROWS, KVD, DIM);
    gemm_bf16_f32<<<dim3(KVD / 128, MROWS / 128), 256, 0, stream>>>(xb, wvb, vf, MROWS, KVD, DIM);

    // 3) RoPE + scatter (also fills K/V f32 cache outputs)
    rope_scatter<<<BATCH * SEQ, 256, 0, stream>>>(qf, kf, vf, cosp, sinp,
                                                  qb16, kb16, vb16, kcache, vcache);

    // 4) causal flash attention
    flash_attn<<<dim3(SEQ / 128, BATCH * NHEAD), 256, 0, stream>>>(qb16, kb16, vb16, ctxb);

    // 5) output projection
    gemm_bf16_f32<<<dim3(DIM / 128, MROWS / 128), 256, 0, stream>>>(ctxb, wob, out, MROWS, DIM, DIM);
}
